// MD_LSTM_56667798503985
// MI455X (gfx1250) — matrix operand order repaired and checked by reading, not compiled, not run on hardware
//
#include <hip/hip_runtime.h>
#include <hip/hip_bf16.h>

// ---------------------------------------------------------------------------
// 2-layer LSTM encoder/decoder for MI455X (gfx1250, wave32, WMMA bf16).
// Host enqueues ~3 kernels per time step (sequential recurrence); GEMMs run
// on v_wmma_f32_16x16x32_bf16 with f32 accumulation. Round 2: M=32 register
// blocking per wave so every weight (B) fragment feeds two WMMAs, halving
// weight fetch traffic (the L2-bandwidth-bound resource of this workload).
// ---------------------------------------------------------------------------

typedef __attribute__((ext_vector_type(16))) __bf16 v16bf;
typedef __attribute__((ext_vector_type(8)))  __bf16 v8bf;
typedef __attribute__((ext_vector_type(8)))  float  v8f;

constexpr int Bsz  = 256;   // batch
constexpr int Ssz  = 100;   // encoder steps
constexpr int Fsz  = 256;   // feature dim
constexpr int Hsz  = 1024;  // hidden dim
constexpr int Tsz  = 200;   // decoder length parameter t
constexpr int OUTS = Ssz + Tsz;  // 300 output time columns

#define DEVFN static __device__ __forceinline__

DEVFN float sigm(float x) { return 1.0f / (1.0f + __expf(-x)); }

// --- WMMA fragment loaders (bf16, 16x16x32 shape) --------------------------
// A matrix 16x32 (MxK), 16-bit layout (ISA 7.12.2):
//   lanes 0-15 : row M=lane,    VGPR0-3 = K 0..7,  VGPR4-7 = K 16..23
//   lanes 16-31: row M=lane-16, VGPR0-3 = K 8..15, VGPR4-7 = K 24..31
DEVFN v16bf load_a16x32(const __bf16* __restrict__ base, int lda,
                        int m0, int k, int lane) {
  const int r = lane & 15, hi = lane >> 4;
  const __bf16* p = base + (size_t)(m0 + r) * (size_t)lda + k;
  v8bf lo = *reinterpret_cast<const v8bf*>(p + hi * 8);
  v8bf hf = *reinterpret_cast<const v8bf*>(p + 16 + hi * 8);
  v16bf o;
#pragma unroll
  for (int i = 0; i < 8; ++i) { o[i] = lo[i]; o[i + 8] = hf[i]; }
  return o;
}

// B matrix 32x16 (KxN). W stored [N_total, K] row-major (GEMM with W^T), so
// column n of B == row (nrow + lane&15) of W, contiguous along K.
//   lanes 0-15 : K 0..7, 16..23 ; lanes 16-31 : K 8..15, 24..31
DEVFN v16bf load_b32x16(const __bf16* __restrict__ w, int ldw,
                        int nrow, int k, int lane) {
  const int c = lane & 15, hi = lane >> 4;
  const __bf16* p = w + (size_t)(nrow + c) * (size_t)ldw + k;
  v8bf lo = *reinterpret_cast<const v8bf*>(p + hi * 8);
  v8bf hf = *reinterpret_cast<const v8bf*>(p + 16 + hi * 8);
  v16bf o;
#pragma unroll
  for (int i = 0; i < 8; ++i) { o[i] = lo[i]; o[i + 8] = hf[i]; }
  return o;
}

DEVFN v8f wmma_bf16(v16bf a, v16bf b, v8f c) {
  return __builtin_amdgcn_wmma_f32_16x16x32_bf16(
      /*neg_a=*/false, a, /*neg_b=*/false, b,
      /*c_mod=*/(short)0, c, /*reuse_a=*/false, /*reuse_b=*/false);
}

// --- Fused LSTM cell: gates GEMM (2 operands) + activations ----------------
// gates[b, g*H+n] = sum_k A1[b,k] W1[g*H+n,k] + sum_k A2[b,k] W2[g*H+n,k] + b
// c_new = sig(f)*c_prev + sig(i)*tanh(g);  h_new = sig(o)*tanh(c_new)
// Wave tile: M=32 (2 sub-tiles) x N=16, all 4 gates in registers
// (8 accumulators). Block 8 waves (2 M x 4 N) => 64x64 tile.
// Grid (B/64, H/64) = (4, 16).
__global__ __launch_bounds__(256)
void lstm_cell_wmma(const __bf16* __restrict__ A1, int lda1, int K1,
                    const __bf16* __restrict__ A2, int lda2, int K2,
                    const __bf16* __restrict__ W1,   // [4H, K1]
                    const __bf16* __restrict__ W2,   // [4H, K2]
                    const float*  __restrict__ bias, // [4H]
                    const float*  __restrict__ c_prev,
                    float*        __restrict__ c_new,
                    __bf16*       __restrict__ h_new) {
  const int lane = threadIdx.x & 31;
  const int wave = threadIdx.x >> 5;
  const int m0 = (blockIdx.x * 2 + (wave & 1)) * 32;   // 32-row wave tile
  const int n0 = (blockIdx.y * 4 + (wave >> 1)) * 16;

  v8f acc[4][2];
#pragma unroll
  for (int g = 0; g < 4; ++g)
#pragma unroll
    for (int mi = 0; mi < 2; ++mi)
      acc[g][mi] = (v8f){0.f,0.f,0.f,0.f,0.f,0.f,0.f,0.f};

  for (int k = 0; k < K1; k += 32) {
    v16bf a0 = load_a16x32(A1, lda1, m0,      k, lane);
    v16bf a1 = load_a16x32(A1, lda1, m0 + 16, k, lane);
#pragma unroll
    for (int g = 0; g < 4; ++g) {
      v16bf b = load_b32x16(W1, K1, g * Hsz + n0, k, lane);
      acc[g][0] = wmma_bf16(a0, b, acc[g][0]);
      acc[g][1] = wmma_bf16(a1, b, acc[g][1]);
    }
  }
  for (int k = 0; k < K2; k += 32) {
    v16bf a0 = load_a16x32(A2, lda2, m0,      k, lane);
    v16bf a1 = load_a16x32(A2, lda2, m0 + 16, k, lane);
#pragma unroll
    for (int g = 0; g < 4; ++g) {
      v16bf b = load_b32x16(W2, K2, g * Hsz + n0, k, lane);
      acc[g][0] = wmma_bf16(a0, b, acc[g][0]);
      acc[g][1] = wmma_bf16(a1, b, acc[g][1]);
    }
  }

  // Epilogue: C/D layout => lane<16: M=j, N=lane ; lane>=16: M=j+8, N=lane-16
  const int n  = lane & 15, hi = lane >> 4;
  const int col = n0 + n;
  const float bi = bias[0 * Hsz + col];
  const float bf = bias[1 * Hsz + col];
  const float bg = bias[2 * Hsz + col];
  const float bo = bias[3 * Hsz + col];
#pragma unroll
  for (int mi = 0; mi < 2; ++mi) {
#pragma unroll
    for (int j = 0; j < 8; ++j) {
      const int row = m0 + mi * 16 + hi * 8 + j;
      const size_t idx = (size_t)row * Hsz + col;
      const float gi = acc[0][mi][j] + bi;
      const float gf = acc[1][mi][j] + bf;
      const float gg = acc[2][mi][j] + bg;
      const float go = acc[3][mi][j] + bo;
      const float cp = c_prev[idx];
      const float cn = sigm(gf) * cp + sigm(gi) * tanhf(gg);
      c_new[idx] = cn;
      h_new[idx] = (__bf16)(sigm(go) * tanhf(cn));
    }
  }
}

// --- Output projection: y = h1 @ out_W^T (+ epilogue variants) -------------
// encoder (dy_std==null): out[row,col] = x[row,col] + y + out_b
// decoder: dxt = dy_std*(y+out_b)+dy_mu ; new = last + dxt ; out = new ;
//          last (f32 and bf16) updated in place.
// Wave tile M=32 x N=16. Block 8 waves (2 M x 4 N). Grid (B/64, F/64)=(4,4).
__global__ __launch_bounds__(256)
void proj_wmma(const __bf16* __restrict__ Hin,     // [B,H] bf16
               const __bf16* __restrict__ Wout,    // [F,H] bf16
               const float*  __restrict__ out_b,   // [F]
               const float*  __restrict__ xrow, int xstride,   // encoder
               float*        __restrict__ outcol, int ostride, // d_out slice
               const float*  __restrict__ dy_std,
               const float*  __restrict__ dy_mu,
               float*        __restrict__ last,    // [B,F] f32 (decoder)
               __bf16*       __restrict__ last_bf) {
  const int lane = threadIdx.x & 31;
  const int wave = threadIdx.x >> 5;
  const int m0 = (blockIdx.x * 2 + (wave & 1)) * 32;
  const int n0 = (blockIdx.y * 4 + (wave >> 1)) * 16;

  v8f acc0 = (v8f){0.f,0.f,0.f,0.f,0.f,0.f,0.f,0.f};
  v8f acc1 = acc0;
  for (int k = 0; k < Hsz; k += 32) {
    v16bf a0 = load_a16x32(Hin, Hsz, m0,      k, lane);
    v16bf a1 = load_a16x32(Hin, Hsz, m0 + 16, k, lane);
    v16bf b  = load_b32x16(Wout, Hsz, n0, k, lane);
    acc0 = wmma_bf16(a0, b, acc0);
    acc1 = wmma_bf16(a1, b, acc1);
  }

  const int n  = lane & 15, hi = lane >> 4;
  const int col = n0 + n;
  const float ob = out_b[col];
  if (dy_std == nullptr) {  // encoder: out = x_t + dx
#pragma unroll
    for (int mi = 0; mi < 2; ++mi) {
      const v8f& acc = mi ? acc1 : acc0;
#pragma unroll
      for (int j = 0; j < 8; ++j) {
        const int row = m0 + mi * 16 + hi * 8 + j;
        const float res = xrow[(size_t)row * xstride + col] + acc[j] + ob;
        outcol[(size_t)row * ostride + col] = res;
      }
    }
  } else {                  // decoder: last += dy_std*(y+b)+dy_mu
    const float s  = dy_std[col];
    const float mu = dy_mu[col];
#pragma unroll
    for (int mi = 0; mi < 2; ++mi) {
      const v8f& acc = mi ? acc1 : acc0;
#pragma unroll
      for (int j = 0; j < 8; ++j) {
        const int row = m0 + mi * 16 + hi * 8 + j;
        const size_t li = (size_t)row * Fsz + col;
        const float dxt = s * (acc[j] + ob) + mu;
        const float res = last[li] + dxt;
        outcol[(size_t)row * ostride + col] = res;
        last[li]    = res;
        last_bf[li] = (__bf16)res;
      }
    }
  }
}

// --- Utilities --------------------------------------------------------------
__global__ void k_cvt_bf16(const float* __restrict__ s, __bf16* __restrict__ d,
                           long long n) {
  long long i = (long long)blockIdx.x * blockDim.x + threadIdx.x;
  const long long st = (long long)gridDim.x * blockDim.x;
  for (; i < n; i += st) d[i] = (__bf16)s[i];
}
__global__ void k_zero_f32(float* p, long long n) {
  long long i = (long long)blockIdx.x * blockDim.x + threadIdx.x;
  const long long st = (long long)gridDim.x * blockDim.x;
  for (; i < n; i += st) p[i] = 0.f;
}
__global__ void k_zero_bf16(__bf16* p, long long n) {
  long long i = (long long)blockIdx.x * blockDim.x + threadIdx.x;
  const long long st = (long long)gridDim.x * blockDim.x;
  for (; i < n; i += st) p[i] = (__bf16)0.f;
}
// out[:,0,:] = x[:,0,:]
__global__ void k_copy_t0(const float* __restrict__ x, float* __restrict__ out) {
  const int i = blockIdx.x * blockDim.x + threadIdx.x;
  if (i < Bsz * Fsz) {
    const int b = i / Fsz, f = i % Fsz;
    out[((size_t)b * OUTS + 0) * Fsz + f] = x[((size_t)b * Ssz + 0) * Fsz + f];
  }
}
// last = out[:, S, :] (f32 + bf16)
__global__ void k_init_last(const float* __restrict__ out,
                            float* __restrict__ last,
                            __bf16* __restrict__ lastbf) {
  const int i = blockIdx.x * blockDim.x + threadIdx.x;
  if (i < Bsz * Fsz) {
    const int b = i / Fsz, f = i % Fsz;
    const float v = out[((size_t)b * OUTS + Ssz) * Fsz + f];
    last[i] = v;
    lastbf[i] = (__bf16)v;
  }
}

extern "C" void kernel_launch(void* const* d_in, const int* in_sizes, int n_in,
                              void* d_out, int out_size, void* d_ws, size_t ws_size,
                              hipStream_t stream) {
  const float* x     = (const float*)d_in[0];   // [B,S,F]
  const float* Wih0  = (const float*)d_in[1];   // [4H,F]
  const float* Whh0  = (const float*)d_in[2];   // [4H,H]
  const float* b0    = (const float*)d_in[3];   // [4H]
  const float* Wih1  = (const float*)d_in[4];   // [4H,H]
  const float* Whh1  = (const float*)d_in[5];   // [4H,H]
  const float* b1    = (const float*)d_in[6];   // [4H]
  const float* outW  = (const float*)d_in[7];   // [F,H]
  const float* outb  = (const float*)d_in[8];   // [F]
  const float* dymu  = (const float*)d_in[9];   // [F]
  const float* dystd = (const float*)d_in[10];  // [F]
  float* out = (float*)d_out;                   // [B, S+T, F]

  // Workspace carve-out (256B aligned regions), ~46 MB total.
  char* ws = (char*)d_ws;
  auto alloc = [&](size_t bytes) -> char* {
    char* p = ws;
    ws += (bytes + 255) & ~(size_t)255;
    return p;
  };
  __bf16* Wih0b = (__bf16*)alloc((size_t)4 * Hsz * Fsz * 2);
  __bf16* Whh0b = (__bf16*)alloc((size_t)4 * Hsz * Hsz * 2);
  __bf16* Wih1b = (__bf16*)alloc((size_t)4 * Hsz * Hsz * 2);
  __bf16* Whh1b = (__bf16*)alloc((size_t)4 * Hsz * Hsz * 2);
  __bf16* outWb = (__bf16*)alloc((size_t)Fsz * Hsz * 2);
  __bf16* Xb    = (__bf16*)alloc((size_t)Bsz * Ssz * Fsz * 2);
  __bf16* h0b[2] = { (__bf16*)alloc((size_t)Bsz * Hsz * 2),
                     (__bf16*)alloc((size_t)Bsz * Hsz * 2) };
  __bf16* h1b[2] = { (__bf16*)alloc((size_t)Bsz * Hsz * 2),
                     (__bf16*)alloc((size_t)Bsz * Hsz * 2) };
  float* c0 = (float*)alloc((size_t)Bsz * Hsz * 4);
  float* c1 = (float*)alloc((size_t)Bsz * Hsz * 4);
  float*  last   = (float*)alloc((size_t)Bsz * Fsz * 4);
  __bf16* lastbf = (__bf16*)alloc((size_t)Bsz * Fsz * 2);

  const dim3 blk(256);
  const dim3 cvtg(2048);
  // bf16 conversions of weights and encoder input (once per call)
  k_cvt_bf16<<<cvtg, blk, 0, stream>>>(Wih0, Wih0b, (long long)4 * Hsz * Fsz);
  k_cvt_bf16<<<cvtg, blk, 0, stream>>>(Whh0, Whh0b, (long long)4 * Hsz * Hsz);
  k_cvt_bf16<<<cvtg, blk, 0, stream>>>(Wih1, Wih1b, (long long)4 * Hsz * Hsz);
  k_cvt_bf16<<<cvtg, blk, 0, stream>>>(Whh1, Whh1b, (long long)4 * Hsz * Hsz);
  k_cvt_bf16<<<cvtg, blk, 0, stream>>>(outW, outWb, (long long)Fsz * Hsz);
  k_cvt_bf16<<<cvtg, blk, 0, stream>>>(x, Xb, (long long)Bsz * Ssz * Fsz);
  // zero initial state
  k_zero_bf16<<<256, blk, 0, stream>>>(h0b[0], (long long)Bsz * Hsz);
  k_zero_bf16<<<256, blk, 0, stream>>>(h0b[1], (long long)Bsz * Hsz);
  k_zero_bf16<<<256, blk, 0, stream>>>(h1b[0], (long long)Bsz * Hsz);
  k_zero_bf16<<<256, blk, 0, stream>>>(h1b[1], (long long)Bsz * Hsz);
  k_zero_f32<<<256, blk, 0, stream>>>(c0, (long long)Bsz * Hsz);
  k_zero_f32<<<256, blk, 0, stream>>>(c1, (long long)Bsz * Hsz);
  k_copy_t0<<<(Bsz * Fsz + 255) / 256, blk, 0, stream>>>(x, out);

  const dim3 cellg(Bsz / 64, Hsz / 64);  // (4,16)
  const dim3 projg(Bsz / 64, Fsz / 64);  // (4,4)

  int step = 0;
  // ---- Encoder: 100 steps ----
  for (int t = 0; t < Ssz; ++t, ++step) {
    const int p = step & 1;
    lstm_cell_wmma<<<cellg, blk, 0, stream>>>(
        Xb + (size_t)t * Fsz, Ssz * Fsz, Fsz,
        h0b[p], Hsz, Hsz,
        Wih0b, Whh0b, b0, c0, c0, h0b[p ^ 1]);
    lstm_cell_wmma<<<cellg, blk, 0, stream>>>(
        h0b[p ^ 1], Hsz, Hsz,
        h1b[p], Hsz, Hsz,
        Wih1b, Whh1b, b1, c1, c1, h1b[p ^ 1]);
    proj_wmma<<<projg, blk, 0, stream>>>(
        h1b[p ^ 1], outWb, outb,
        x + (size_t)t * Fsz, Ssz * Fsz,
        out + (size_t)(t + 1) * Fsz, OUTS * Fsz,
        nullptr, nullptr, nullptr, nullptr);
  }

  k_init_last<<<(Bsz * Fsz + 255) / 256, blk, 0, stream>>>(out, last, lastbf);

  // ---- Decoder: t-1 = 199 steps ----
  for (int j = 0; j < Tsz - 1; ++j, ++step) {
    const int p = step & 1;
    lstm_cell_wmma<<<cellg, blk, 0, stream>>>(
        lastbf, Fsz, Fsz,
        h0b[p], Hsz, Hsz,
        Wih0b, Whh0b, b0, c0, c0, h0b[p ^ 1]);
    lstm_cell_wmma<<<cellg, blk, 0, stream>>>(
        h0b[p ^ 1], Hsz, Hsz,
        h1b[p], Hsz, Hsz,
        Wih1b, Whh1b, b1, c1, c1, h1b[p ^ 1]);
    proj_wmma<<<projg, blk, 0, stream>>>(
        h1b[p ^ 1], outWb, outb,
        nullptr, 0,
        out + (size_t)(Ssz + 1 + j) * Fsz, OUTS * Fsz,
        dystd, dymu, last, lastbf);
  }
}
